// Int8Linear_28690381537822
// MI455X (gfx1250) — matrix operand order repaired and checked by reading, not compiled, not run on hardware
//
#include <hip/hip_runtime.h>

// ---------------------------------------------------------------------------
// Int8-weight linear (x fp32 [8192x4096] @ w_int8^T -> [8192x11008]) gfx1250.
//
// Fast path (needs ~150MB workspace):
//   1) prepass: W int8 -> bf16 (exact bit-trick) and X fp32 -> bf16 into ws
//   2) GEMM: triple-buffered LDS stages filled with GLOBAL_LOAD_ASYNC_TO_LDS
//      (GVS addressing, ASYNCcnt-tracked, 2 batches in flight), fragments via
//      ds_load_b128, v_wmma_f32_16x16x32_bf16, f32 epilogue with per-channel
//      scale + bias.
// Fallback (small ws): in-loop conversion kernel.
// ---------------------------------------------------------------------------

typedef __bf16 bf16_t;
typedef bf16_t       v16bf __attribute__((ext_vector_type(16)));
typedef float        v8f   __attribute__((ext_vector_type(8)));
typedef float        vf4   __attribute__((ext_vector_type(4)));
typedef int          vi4   __attribute__((ext_vector_type(4)));
typedef unsigned int vu4   __attribute__((ext_vector_type(4)));
typedef unsigned int vu2   __attribute__((ext_vector_type(2)));

namespace {
constexpr int kM = 8192;        // B*S
constexpr int kN = 11008;       // OUT
constexpr int kK = 4096;        // IN
constexpr int BM = 128;
constexpr int BN = 128;
constexpr int BK = 32;          // K per LDS stage (one bf16 WMMA K-slab)
constexpr int LDSS = BK + 8;    // LDS row stride in bf16 elems (16B pad)
constexpr int NKT = kK / BK;    // 128 k-iterations
constexpr int A_ELEMS = BM * LDSS;            // 5120 elems / 10240 B
constexpr int B_ELEMS = BN * LDSS;            // 5120 elems / 10240 B
constexpr int STAGE_ELEMS = A_ELEMS + B_ELEMS;            // 10240 elems
constexpr unsigned A_BYTES     = A_ELEMS * 2u;            // 10240
constexpr unsigned STAGE_BYTES = STAGE_ELEMS * 2u;        // 20480
}

union FragU {
    v16bf v;
    vf4   f[2];
};

// ---------------------------------------------------------------------------
// CDNA5 async copy global->LDS, GVS mode: mem = SGPR base + VGPR off32.
// 16B per lane, tracked by ASYNCcnt (in-order completion).
// ---------------------------------------------------------------------------
__device__ __forceinline__ void async_copy_b128(unsigned lds_off, unsigned goff,
                                                const bf16_t* sbase) {
    asm volatile("global_load_async_to_lds_b128 %0, %1, %2"
                 :: "v"(lds_off), "v"(goff), "s"(sbase)
                 : "memory");
}
__device__ __forceinline__ void wait_async_le4() {
    asm volatile("s_wait_asynccnt 0x4" ::: "memory");
}
__device__ __forceinline__ void wait_async0() {
    asm volatile("s_wait_asynccnt 0x0" ::: "memory");
}

// ---------------------------------------------------------------------------
// Pre-pass: W int8 -> bf16. Exact: |w| <= 128 has <=8 significant bits, so
// bf16 = high 16 bits of the f32 image (no rounding needed).
// ---------------------------------------------------------------------------
__global__ __launch_bounds__(256)
void convert_w_bf16(const signed char* __restrict__ W, unsigned short* __restrict__ Wb) {
    const size_t i = (size_t)blockIdx.x * 256 + threadIdx.x;   // int4 id
    const vi4 v = reinterpret_cast<const vi4*>(W)[i];
    const int wv[4] = {v.x, v.y, v.z, v.w};
    __align__(16) unsigned short h[16];
#pragma unroll
    for (int a = 0; a < 4; ++a)
#pragma unroll
        for (int b = 0; b < 4; ++b) {
            const float f = (float)(signed char)(wv[a] >> (8 * b));  // v_cvt_f32_i32
            h[a * 4 + b] = (unsigned short)(__float_as_uint(f) >> 16);
        }
    *reinterpret_cast<vu4*>(Wb + i * 16)     = *reinterpret_cast<const vu4*>(h);
    *reinterpret_cast<vu4*>(Wb + i * 16 + 8) = *reinterpret_cast<const vu4*>(h + 8);
}

// Pre-pass: X fp32 -> bf16 (RNE via hardware cvt).
__global__ __launch_bounds__(256)
void convert_x_bf16(const float* __restrict__ X, bf16_t* __restrict__ Xb) {
    const size_t i = (size_t)blockIdx.x * 256 + threadIdx.x;   // float4 id
    const vf4 v = reinterpret_cast<const vf4*>(X)[i];
    __align__(8) bf16_t tmp[4] = {(bf16_t)v.x, (bf16_t)v.y, (bf16_t)v.z, (bf16_t)v.w};
    *reinterpret_cast<vu2*>(Xb + i * 4) = *reinterpret_cast<const vu2*>(tmp);
}

// ---------------------------------------------------------------------------
// Main GEMM: triple-buffered async pipeline + WMMA.
// ---------------------------------------------------------------------------
__global__ __launch_bounds__(256)
void gemm_bf16_wmma_async(const bf16_t* __restrict__ Xb,
                          const bf16_t* __restrict__ Wb,
                          const float* __restrict__ scale,
                          const float* __restrict__ bias,
                          float* __restrict__ out)
{
    // 3 stages x [A tile][B tile], 61440 B total.
    __shared__ __align__(16) bf16_t smem[3 * STAGE_ELEMS];

    const int t     = threadIdx.x;       // 8 waves of 32
    const int lane  = t & 31;
    const int lr    = lane & 15;
    const int hsel  = lane >> 4;
    const int wave  = t >> 5;
    const int waveM = wave >> 1;         // 0..3 -> 32-row band
    const int waveN = wave & 1;          // 0..1 -> 64-col band

    const int m0 = blockIdx.y * BM;
    const int n0 = blockIdx.x * BN;

    // Copy assignment: 16B chunk id = t + i*256 (i=0,1) per operand;
    // row = id>>2 (0..127), chunk-in-row = id&3 (32 bf16 = 4 x 16B per row).
    const int cr = t >> 2;               // rows 0..63 (+64 for i=1)
    const int cc = t & 3;

    // Uniform SGPR bases (advance with kt); loop-invariant per-lane offsets.
    const bf16_t* baseA = Xb + (size_t)m0 * kK;
    const bf16_t* baseB = Wb + (size_t)n0 * kK;
    unsigned gOffA[2], gOffB[2], ldsA[2], ldsB[2];
#pragma unroll
    for (int i = 0; i < 2; ++i) {
        const int row = cr + i * 64;
        gOffA[i] = (unsigned)(row * kK + cc * 8) * 2u;                 // bytes
        gOffB[i] = gOffA[i];
        ldsA[i]  = (unsigned)(row * LDSS + cc * 8) * 2u;               // in-stage
        ldsB[i]  = A_BYTES + ldsA[i];
    }
    const unsigned smemBase = (unsigned)(unsigned long long)(void*)smem;

    auto issueCopies = [&](int kt, int stage) {
        const unsigned soff = smemBase + (unsigned)stage * STAGE_BYTES;
        const bf16_t* bA = baseA + (size_t)kt * BK;
        const bf16_t* bB = baseB + (size_t)kt * BK;
#pragma unroll
        for (int i = 0; i < 2; ++i) {
            async_copy_b128(soff + ldsA[i], gOffA[i], bA);
            async_copy_b128(soff + ldsB[i], gOffB[i], bB);
        }
    };

    v8f acc[2][4];
    const v8f zero = {0.f, 0.f, 0.f, 0.f, 0.f, 0.f, 0.f, 0.f};
#pragma unroll
    for (int mi = 0; mi < 2; ++mi)
#pragma unroll
        for (int ni = 0; ni < 4; ++ni)
            acc[mi][ni] = zero;

    auto computeStage = [&](int stage) {
        const bf16_t* Ab = smem + stage * STAGE_ELEMS;
        const bf16_t* Bb = Ab + A_ELEMS;
        // Fragments per ISA 7.12.2 bf16 layouts.
        FragU fa[2];
#pragma unroll
        for (int mi = 0; mi < 2; ++mi) {
            const bf16_t* ap = Ab + (waveM * 32 + mi * 16 + lr) * LDSS + hsel * 8;
            fa[mi].f[0] = *reinterpret_cast<const vf4*>(ap);
            fa[mi].f[1] = *reinterpret_cast<const vf4*>(ap + 16);
        }
        FragU fb[4];
#pragma unroll
        for (int ni = 0; ni < 4; ++ni) {
            const bf16_t* bp = Bb + (waveN * 64 + ni * 16 + lr) * LDSS + hsel * 8;
            fb[ni].f[0] = *reinterpret_cast<const vf4*>(bp);
            fb[ni].f[1] = *reinterpret_cast<const vf4*>(bp + 16);
        }
#pragma unroll
        for (int mi = 0; mi < 2; ++mi)
#pragma unroll
            for (int ni = 0; ni < 4; ++ni)
                acc[mi][ni] = __builtin_amdgcn_wmma_f32_16x16x32_bf16(
                    false, fa[mi].v, false, fb[ni].v,
                    (short)0, acc[mi][ni], false, false);
    };

    // Pipeline: stages kt, kt+1 in flight entering iteration kt.
    issueCopies(0, 0);
    issueCopies(1, 1);

    int bc = 0;   // compute stage index = kt % 3
#pragma unroll 1
    for (int kt = 0; kt < NKT - 2; ++kt) {
        wait_async_le4();     // batch kt complete (in-order), kt+1 still flying
        __syncthreads();      // kt data visible; stage (kt-1)%3 free to refill
        issueCopies(kt + 2, bc ? bc - 1 : 2);   // (bc+2)%3
        computeStage(bc);
        bc = (bc == 2) ? 0 : bc + 1;
    }
    // kt = NKT-2: batches NKT-2, NKT-1 outstanding.
    wait_async_le4();
    __syncthreads();
    computeStage(bc);
    bc = (bc == 2) ? 0 : bc + 1;
    // kt = NKT-1: last batch.
    wait_async0();
    __syncthreads();
    computeStage(bc);

    // Epilogue: per-channel dequant scale + bias.
    // C/D layout: VGPR r -> (M = r + 8*hsel, N = lr) within each 16x16 tile.
#pragma unroll
    for (int ni = 0; ni < 4; ++ni) {
        const int col = n0 + waveN * 64 + ni * 16 + lr;
        const float s  = scale[col];
        const float bz = bias[col];
#pragma unroll
        for (int mi = 0; mi < 2; ++mi) {
            const int row0 = m0 + waveM * 32 + mi * 16 + hsel * 8;
#pragma unroll
            for (int r = 0; r < 8; ++r)
                out[(size_t)(row0 + r) * kN + col] = acc[mi][ni][r] * s + bz;
        }
    }
}

// ---------------------------------------------------------------------------
// Fallback (small workspace): conversions in-loop, double-buffered LDS.
// ---------------------------------------------------------------------------
__global__ __launch_bounds__(256)
void int8_linear_bf16_wmma(const float* __restrict__ X,
                           const signed char* __restrict__ W,
                           const float* __restrict__ scale,
                           const float* __restrict__ bias,
                           float* __restrict__ out)
{
    __shared__ __align__(16) bf16_t As[2][BM * LDSS];
    __shared__ __align__(16) bf16_t Bs[2][BN * LDSS];

    const int t     = threadIdx.x;
    const int lane  = t & 31;
    const int lr    = lane & 15;
    const int hsel  = lane >> 4;
    const int wave  = t >> 5;
    const int waveM = wave >> 1;
    const int waveN = wave & 1;

    const int m0 = blockIdx.y * BM;
    const int n0 = blockIdx.x * BN;

    const int bRow = t >> 1;
    const int bKo  = (t & 1) * 16;

    vf4 aReg[4];
    vi4 bReg;

    auto loadTile = [&](int kt) {
        const size_t kbase = (size_t)kt * BK;
#pragma unroll
        for (int i = 0; i < 4; ++i) {
            const int id  = t + i * 256;
            const int row = id >> 3;
            const int kq  = id & 7;
            aReg[i] = *reinterpret_cast<const vf4*>(
                X + (size_t)(m0 + row) * kK + kbase + kq * 4);
        }
        bReg = *reinterpret_cast<const vi4*>(
            W + (size_t)(n0 + bRow) * kK + kbase + bKo);
    };

    auto storeTile = [&](int buf) {
#pragma unroll
        for (int i = 0; i < 4; ++i) {
            const int id  = t + i * 256;
            const int row = id >> 3;
            const int kq  = id & 7;
            __align__(8) bf16_t tmp[4];
            tmp[0] = (bf16_t)aReg[i].x;
            tmp[1] = (bf16_t)aReg[i].y;
            tmp[2] = (bf16_t)aReg[i].z;
            tmp[3] = (bf16_t)aReg[i].w;
            *reinterpret_cast<vu2*>(&As[buf][row * LDSS + kq * 4]) =
                *reinterpret_cast<const vu2*>(tmp);
        }
        __align__(16) unsigned short wt[16];
        const int wv[4] = {bReg.x, bReg.y, bReg.z, bReg.w};
#pragma unroll
        for (int i = 0; i < 4; ++i)
#pragma unroll
            for (int j = 0; j < 4; ++j) {
                const float f = (float)(signed char)(wv[i] >> (8 * j));
                wt[i * 4 + j] = (unsigned short)(__float_as_uint(f) >> 16); // exact
            }
        bf16_t* dst = &Bs[buf][bRow * LDSS + bKo];
        *reinterpret_cast<vu4*>(dst)     = *reinterpret_cast<const vu4*>(wt);
        *reinterpret_cast<vu4*>(dst + 8) = *reinterpret_cast<const vu4*>(wt + 8);
    };

    v8f acc[2][4];
    const v8f zero = {0.f, 0.f, 0.f, 0.f, 0.f, 0.f, 0.f, 0.f};
#pragma unroll
    for (int mi = 0; mi < 2; ++mi)
#pragma unroll
        for (int ni = 0; ni < 4; ++ni)
            acc[mi][ni] = zero;

    loadTile(0);
    storeTile(0);
    __syncthreads();

    int buf = 0;
#pragma unroll 1
    for (int kt = 0; kt < NKT; ++kt) {
        const bool more = (kt + 1) < NKT;
        if (more) loadTile(kt + 1);

        FragU fa[2];
#pragma unroll
        for (int mi = 0; mi < 2; ++mi) {
            const bf16_t* ap =
                &As[buf][(waveM * 32 + mi * 16 + lr) * LDSS + hsel * 8];
            fa[mi].f[0] = *reinterpret_cast<const vf4*>(ap);
            fa[mi].f[1] = *reinterpret_cast<const vf4*>(ap + 16);
        }
        FragU fb[4];
#pragma unroll
        for (int ni = 0; ni < 4; ++ni) {
            const bf16_t* bp =
                &Bs[buf][(waveN * 64 + ni * 16 + lr) * LDSS + hsel * 8];
            fb[ni].f[0] = *reinterpret_cast<const vf4*>(bp);
            fb[ni].f[1] = *reinterpret_cast<const vf4*>(bp + 16);
        }

#pragma unroll
        for (int mi = 0; mi < 2; ++mi)
#pragma unroll
            for (int ni = 0; ni < 4; ++ni)
                acc[mi][ni] = __builtin_amdgcn_wmma_f32_16x16x32_bf16(
                    false, fa[mi].v, false, fb[ni].v,
                    (short)0, acc[mi][ni], false, false);

        if (more) storeTile(buf ^ 1);
        __syncthreads();
        buf ^= 1;
    }

#pragma unroll
    for (int ni = 0; ni < 4; ++ni) {
        const int col = n0 + waveN * 64 + ni * 16 + lr;
        const float s  = scale[col];
        const float bz = bias[col];
#pragma unroll
        for (int mi = 0; mi < 2; ++mi) {
            const int row0 = m0 + waveM * 32 + mi * 16 + hsel * 8;
#pragma unroll
            for (int r = 0; r < 8; ++r)
                out[(size_t)(row0 + r) * kN + col] = acc[mi][ni][r] * s + bz;
        }
    }
}

extern "C" void kernel_launch(void* const* d_in, const int* in_sizes, int n_in,
                              void* d_out, int out_size, void* d_ws, size_t ws_size,
                              hipStream_t stream) {
    (void)in_sizes; (void)n_in; (void)out_size;
    const float*       x  = (const float*)d_in[0];
    const signed char* w  = (const signed char*)d_in[1];
    const float*       sc = (const float*)d_in[2];
    const float*       bi = (const float*)d_in[3];
    float*             o  = (float*)d_out;

    const size_t needW = (size_t)kN * kK * sizeof(unsigned short); // 90,177,536
    const size_t needX = (size_t)kM * kK * sizeof(unsigned short); // 67,108,864
    const dim3 grid(kN / BN, kM / BM);                             // 86 x 64

    if (ws_size >= needW + needX) {
        unsigned short* Wb = (unsigned short*)d_ws;
        bf16_t*         Xb = (bf16_t*)((char*)d_ws + needW);
        convert_w_bf16<<<(kN * kK) / (16 * 256), 256, 0, stream>>>(w, Wb);
        convert_x_bf16<<<(kM * kK) / (4 * 256), 256, 0, stream>>>(x, Xb);
        gemm_bf16_wmma_async<<<grid, 256, 0, stream>>>(
            Xb, (const bf16_t*)Wb, sc, bi, o);
    } else {
        int8_linear_bf16_wmma<<<grid, 256, 0, stream>>>(x, w, sc, bi, o);
    }
}
